// FlorenceVisionEncoder_1460288881462
// MI455X (gfx1250) — matrix operand order repaired and checked by reading, not compiled, not run on hardware
//
#include <hip/hip_runtime.h>
#include <hip/hip_bf16.h>

// ---------------- types ----------------
typedef _Float16 v16h __attribute__((ext_vector_type(16)));
typedef _Float16 half8 __attribute__((ext_vector_type(8)));
typedef float    v8f  __attribute__((ext_vector_type(8)));
typedef unsigned int u32x4 __attribute__((ext_vector_type(4)));
typedef int          i32x4 __attribute__((ext_vector_type(4)));
typedef int          i32x8 __attribute__((ext_vector_type(8)));

#define DIM_B 4
#define DIM_N 1024          // tokens (= 32*32 window)
#define DIM_D 384
#define DIM_NH 12
#define DIM_HD 32
#define DIM_DEPTH 6
#define DIM_TABLE 3969      // (2*32-1)^2
#define ROWS (DIM_B * DIM_N)   // 4096

#if defined(__has_builtin)
#if __has_builtin(__builtin_amdgcn_tensor_load_to_lds) && __has_builtin(__builtin_amdgcn_s_wait_tensorcnt)
#define USE_TDM 1
#endif
#endif
#ifndef USE_TDM
#define USE_TDM 0
#endif

// ---------------- fragment helpers (wave32 WMMA f16 layouts) ----------------
__device__ __forceinline__ v16h frag_combine(half8 lo, half8 hi) {
  v16h f;
#pragma unroll
  for (int j = 0; j < 8; ++j) { f[j] = lo[j]; f[8 + j] = hi[j]; }
  return f;
}

// A matrix 16x32 f16: lane r=lane&15 holds row r; half=lane>>4 selects
// K chunks [half*8 .. +7] and [16+half*8 .. +7]
__device__ __forceinline__ v16h load_frag_a(const _Float16* p, int stride, int lane) {
  int r = lane & 15, h = lane >> 4;
  const half8* p0 = (const half8*)(p + r * stride + h * 8);
  const half8* p1 = (const half8*)(p + r * stride + 16 + h * 8);
  return frag_combine(*p0, *p1);
}

// B matrix 32x16 f16 from row-major [col][K]: lane c=lane&15 is column c,
// half=lane>>4 selects K chunks [half*8 .. +7] and [16+half*8 .. +7]
__device__ __forceinline__ v16h load_frag_b(const _Float16* p, int stride, int lane) {
  int c = lane & 15, h = lane >> 4;
  const half8* p0 = (const half8*)(p + c * stride + h * 8);
  const half8* p1 = (const half8*)(p + c * stride + 16 + h * 8);
  return frag_combine(*p0, *p1);
}

__device__ __forceinline__ float gelu_exact(float x) {
  return 0.5f * x * (1.0f + erff(x * 0.70710678118654752440f));
}

#if USE_TDM
// Issue one TDM 2-D tile load: [tile_y rows x tile_x f16] from row-major
// global (row stride = stride_elems f16) into contiguous LDS at lds_off.
__device__ __forceinline__ void tdm_load_2d_f16(
    unsigned lds_off, const _Float16* gptr,
    unsigned tile_x, unsigned tile_y,
    unsigned tensor_x, unsigned tensor_y,
    unsigned long long stride_elems) {
  unsigned long long ga = (unsigned long long)(uintptr_t)gptr;
  u32x4 g0;
  g0[0] = 1u;                                   // count=1 (valid), user mode
  g0[1] = lds_off;                              // lds_addr        [63:32]
  g0[2] = (unsigned)(ga & 0xFFFFFFFFu);         // global_addr lo  [95:64]
  g0[3] = (unsigned)((ga >> 32) & 0x01FFFFFFu)  // global_addr hi  [120:96]
          | (2u << 30);                         // type=2 (image)  [127:126]
  i32x8 g1;
  g1[0] = (int)(1u << 16);                      // data_size=1 -> 2 bytes
  g1[1] = (int)((tensor_x & 0xFFFFu) << 16);    // tensor_dim0 lo16 @ [63:48]
  g1[2] = (int)(((tensor_x >> 16) & 0xFFFFu) |  // tensor_dim0 hi16 @ [79:64]
                ((tensor_y & 0xFFFFu) << 16));  // tensor_dim1 lo16 @ [95:80]
  g1[3] = (int)(((tensor_y >> 16) & 0xFFFFu) |  // tensor_dim1 hi16 @ [111:96]
                ((tile_x & 0xFFFFu) << 16));    // tile_dim0        @ [127:112]
  g1[4] = (int)(tile_y & 0xFFFFu);              // tile_dim1 @[143:128], tile_dim2=0
  g1[5] = (int)(unsigned)(stride_elems & 0xFFFFFFFFu);         // stride0 lo32
  g1[6] = (int)(unsigned)((stride_elems >> 32) & 0xFFFFu);     // stride0 hi16
  g1[7] = 0;
  i32x4 z4; z4[0] = z4[1] = z4[2] = z4[3] = 0;
#if __clang_major__ >= 23
  i32x8 z8;
#pragma unroll
  for (int j = 0; j < 8; ++j) z8[j] = 0;
  __builtin_amdgcn_tensor_load_to_lds(g0, g1, z4, z4, z8, 0);
#else
  __builtin_amdgcn_tensor_load_to_lds(g0, g1, z4, z4, 0);
#endif
}
#endif

// ---------------- fp32 -> f16 convert ----------------
__global__ void cvt_f16_kernel(const float* __restrict__ in, _Float16* __restrict__ out, int n) {
  int i = blockIdx.x * 256 + threadIdx.x;
  if (i < n) out[i] = (_Float16)in[i];
}

// ---------------- patch embed: 4x4/4 conv, NCHW -> [B*N, D] ----------------
__global__ __launch_bounds__(128) void patch_embed_kernel(
    const float* __restrict__ px, const float* __restrict__ cw,
    const float* __restrict__ cb, float* __restrict__ x) {
  int bn = blockIdx.x;            // b*1024 + n
  int b = bn >> 10, n = bn & 1023;
  int gy = n >> 5, gx = n & 31;
  __shared__ float spx[48];
  int tid = threadIdx.x;
  if (tid < 48) {
    int c = tid >> 4, ky = (tid >> 2) & 3, kx = tid & 3;
    spx[tid] = px[(((size_t)(b * 3 + c) * 128) + gy * 4 + ky) * 128 + gx * 4 + kx];
  }
  __syncthreads();
  for (int d = tid; d < DIM_D; d += 128) {
    float acc = cb[d];
    const float* w = cw + (size_t)d * 48;
#pragma unroll
    for (int j = 0; j < 48; ++j) acc += spx[j] * w[j];
    x[(size_t)bn * DIM_D + d] = acc;
  }
}

// ---------------- LayerNorm over D=384, one row per block ----------------
__device__ __forceinline__ float block_sum_384(float v, float* red, int tid) {
  int lane = tid & 31, wid = tid >> 5;
#pragma unroll
  for (int m = 16; m; m >>= 1) v += __shfl_xor(v, m);
  if (lane == 0) red[wid] = v;
  __syncthreads();
  v = red[0] + red[1] + red[2] + red[3];
  __syncthreads();
  return v;
}

template <typename T>
__global__ __launch_bounds__(128) void ln_kernel(
    const float* __restrict__ x, const float* __restrict__ g,
    const float* __restrict__ bt, T* __restrict__ out) {
  __shared__ float red[4];
  int row = blockIdx.x, tid = threadIdx.x;
  const float* xr = x + (size_t)row * DIM_D;
  float v0 = xr[tid], v1 = xr[tid + 128], v2 = xr[tid + 256];
  float mean = block_sum_384(v0 + v1 + v2, red, tid) * (1.0f / 384.0f);
  float d0 = v0 - mean, d1 = v1 - mean, d2 = v2 - mean;
  float var = block_sum_384(d0 * d0 + d1 * d1 + d2 * d2, red, tid) * (1.0f / 384.0f);
  float rstd = rsqrtf(var + 1e-5f);
  T* orow = out + (size_t)row * DIM_D;
  orow[tid]       = (T)(d0 * rstd * g[tid]       + bt[tid]);
  orow[tid + 128] = (T)(d1 * rstd * g[tid + 128] + bt[tid + 128]);
  orow[tid + 256] = (T)(d2 * rstd * g[tid + 256] + bt[tid + 256]);
}

// ---------------- WMMA GEMM: out[M,N] = A[M,K] @ W[N,K]^T (+epilogue) ------
// TDM double-buffered LDS staging: wave 0 issues tensor_load_to_lds for
// tile k+1 while all 8 waves run WMMA on tile k; gated by s_wait_tensorcnt.
// MODE 0: +bias, q-scale, scatter to q/k/v [B,NH,N,HD] f16
// MODE 1: +bias +residual(f32) -> f32 out
// MODE 2: +bias, exact GELU -> f16 out
template <int MODE>
__global__ __launch_bounds__(256) void gemm_kernel(
    const _Float16* __restrict__ A, const _Float16* __restrict__ W,
    const float* __restrict__ bias, const float* __restrict__ res,
    float* __restrict__ outf, _Float16* __restrict__ outh,
    _Float16* __restrict__ qb, _Float16* __restrict__ kb, _Float16* __restrict__ vb,
    int M, int K, int Nout, float qscale) {
  __shared__ _Float16 sA[2][128 * 32];
  __shared__ _Float16 sB[2][128 * 32];
  int tid = threadIdx.x;
  int lane = tid & 31, wid = tid >> 5;
  int wm = wid & 3, wn = wid >> 2;          // 4 waves in M, 2 in N
  int bm = blockIdx.y * 128, bn = blockIdx.x * 128;

  v8f acc[2][4];
#pragma unroll
  for (int mi = 0; mi < 2; ++mi)
#pragma unroll
    for (int ni = 0; ni < 4; ++ni)
#pragma unroll
      for (int j = 0; j < 8; ++j) acc[mi][ni][j] = 0.0f;

  int nk = K >> 5;                          // K / 32 tiles
#if USE_TDM
  unsigned ldsA0 = (unsigned)(uintptr_t)(void*)&sA[0][0];
  unsigned ldsA1 = (unsigned)(uintptr_t)(void*)&sA[1][0];
  unsigned ldsB0 = (unsigned)(uintptr_t)(void*)&sB[0][0];
  unsigned ldsB1 = (unsigned)(uintptr_t)(void*)&sB[1][0];
  if (wid == 0) {
    tdm_load_2d_f16(ldsA0, A + (size_t)bm * K, 32, 128, (unsigned)K, 128, (unsigned long long)K);
    tdm_load_2d_f16(ldsB0, W + (size_t)bn * K, 32, 128, (unsigned)K, 128, (unsigned long long)K);
  }
  for (int kk = 0; kk < nk; ++kk) {
    int buf = kk & 1;
    if (wid == 0) {
      if (kk + 1 < nk) {
        int k1 = (kk + 1) << 5;
        tdm_load_2d_f16(buf ? ldsA0 : ldsA1, A + (size_t)bm * K + k1,
                        32, 128, (unsigned)(K - k1), 128, (unsigned long long)K);
        tdm_load_2d_f16(buf ? ldsB0 : ldsB1, W + (size_t)bn * K + k1,
                        32, 128, (unsigned)(K - k1), 128, (unsigned long long)K);
        __builtin_amdgcn_s_wait_tensorcnt(2);   // tile kk complete (in-order)
      } else {
        __builtin_amdgcn_s_wait_tensorcnt(0);
      }
    }
    __syncthreads();
    v16h af[2], bf[4];
#pragma unroll
    for (int mi = 0; mi < 2; ++mi)
      af[mi] = load_frag_a(&sA[buf][0] + (wm * 32 + mi * 16) * 32, 32, lane);
#pragma unroll
    for (int ni = 0; ni < 4; ++ni)
      bf[ni] = load_frag_b(&sB[buf][0] + (wn * 64 + ni * 16) * 32, 32, lane);
#pragma unroll
    for (int mi = 0; mi < 2; ++mi)
#pragma unroll
      for (int ni = 0; ni < 4; ++ni)
        acc[mi][ni] = __builtin_amdgcn_wmma_f32_16x16x32_f16(
            false, af[mi], false, bf[ni], (short)0, acc[mi][ni], false, false);
    __syncthreads();
  }
#else
  int srow = tid >> 1, sseg = (tid & 1) << 4;
  for (int kk = 0; kk < nk; ++kk) {
    int k0 = kk << 5;
    const half8* sa = (const half8*)(A + (size_t)(bm + srow) * K + k0 + sseg);
    half8* da = (half8*)(&sA[0][0] + srow * 32 + sseg);
    da[0] = sa[0]; da[1] = sa[1];
    const half8* sw = (const half8*)(W + (size_t)(bn + srow) * K + k0 + sseg);
    half8* dw = (half8*)(&sB[0][0] + srow * 32 + sseg);
    dw[0] = sw[0]; dw[1] = sw[1];
    __syncthreads();
    v16h af[2], bf[4];
#pragma unroll
    for (int mi = 0; mi < 2; ++mi)
      af[mi] = load_frag_a(&sA[0][0] + (wm * 32 + mi * 16) * 32, 32, lane);
#pragma unroll
    for (int ni = 0; ni < 4; ++ni)
      bf[ni] = load_frag_b(&sB[0][0] + (wn * 64 + ni * 16) * 32, 32, lane);
#pragma unroll
    for (int mi = 0; mi < 2; ++mi)
#pragma unroll
      for (int ni = 0; ni < 4; ++ni)
        acc[mi][ni] = __builtin_amdgcn_wmma_f32_16x16x32_f16(
            false, af[mi], false, bf[ni], (short)0, acc[mi][ni], false, false);
    __syncthreads();
  }
#endif

  int half = lane >> 4, lc = lane & 15;
  int rbase = bm + wm * 32, cbase = bn + wn * 64;
#pragma unroll
  for (int mi = 0; mi < 2; ++mi)
#pragma unroll
    for (int ni = 0; ni < 4; ++ni)
#pragma unroll
      for (int i = 0; i < 8; ++i) {
        int row = rbase + mi * 16 + half * 8 + i;
        int col = cbase + ni * 16 + lc;
        float val = acc[mi][ni][i] + bias[col];
        if (MODE == 0) {
          int s = col / DIM_D, rem = col % DIM_D;
          int h = rem >> 5, dd = rem & 31;
          int b = row >> 10, n = row & 1023;
          size_t off = (((size_t)(b * DIM_NH + h) * DIM_N) + n) * DIM_HD + dd;
          if (s == 0)      qb[off] = (_Float16)(val * qscale);
          else if (s == 1) kb[off] = (_Float16)val;
          else             vb[off] = (_Float16)val;
        } else if (MODE == 1) {
          size_t off = (size_t)row * Nout + col;
          outf[off] = val + res[off];
        } else {
          outh[(size_t)row * Nout + col] = (_Float16)gelu_exact(val);
        }
      }
}

// ---------------- flash attention, one (b,h,64-row q chunk) per block ------
__global__ __launch_bounds__(128) void attn_kernel(
    const _Float16* __restrict__ q, const _Float16* __restrict__ k,
    const _Float16* __restrict__ v, const float* __restrict__ rb,
    _Float16* __restrict__ out) {
  int qc = blockIdx.x, h = blockIdx.y, b = blockIdx.z;
  int tid = threadIdx.x, lane = tid & 31, wid = tid >> 5;
  int half = lane >> 4, lc = lane & 15;
  size_t base = (size_t)(b * DIM_NH + h) * DIM_N;   // row offset into q/k/v
  int qrow0 = qc * 64 + wid * 16;

  __shared__ _Float16 sVt[32 * 128];        // V^T chunk: [d][key_local]
  __shared__ _Float16 sP[4][16 * 128];      // per-wave probabilities

  v16h qf = load_frag_a(q + (base + qrow0) * DIM_HD, DIM_HD, lane);

  v8f accO[2];
  float m_run[8], l_run[8];
#pragma unroll
  for (int i = 0; i < 8; ++i) { m_run[i] = -1e30f; l_run[i] = 0.0f; accO[0][i] = 0.0f; accO[1][i] = 0.0f; }
  v8f zeroc;
#pragma unroll
  for (int i = 0; i < 8; ++i) zeroc[i] = 0.0f;

  for (int kb0 = 0; kb0 < DIM_N; kb0 += 128) {
    // stage V^T: thread t owns key row kb0+t
    {
      const _Float16* vr = v + (base + kb0 + tid) * DIM_HD;
#pragma unroll
      for (int d = 0; d < 32; ++d) sVt[d * 128 + tid] = vr[d];
    }
    __syncthreads();

    // S = q @ k^T for 8 column tiles of 16 keys
    v8f s[8];
#pragma unroll
    for (int nt = 0; nt < 8; ++nt) {
      v16h bf = load_frag_b(k + (base + kb0 + nt * 16) * DIM_HD, DIM_HD, lane);
      s[nt] = __builtin_amdgcn_wmma_f32_16x16x32_f16(false, qf, false, bf,
                                                     (short)0, zeroc, false, false);
    }

    // + relative position bias, online softmax
    float rmax[8];
#pragma unroll
    for (int i = 0; i < 8; ++i) rmax[i] = -1e30f;
#pragma unroll
    for (int nt = 0; nt < 8; ++nt) {
      int kj = kb0 + nt * 16 + lc;
      int yj = kj >> 5, xj = kj & 31;
#pragma unroll
      for (int i = 0; i < 8; ++i) {
        int qi = qrow0 + i + 8 * half;
        int idx = ((qi >> 5) - yj + 31) * 63 + ((qi & 31) - xj + 31);
        float sv = s[nt][i] + rb[(size_t)idx * DIM_NH + h];
        s[nt][i] = sv;
        rmax[i] = fmaxf(rmax[i], sv);
      }
    }
#pragma unroll
    for (int i = 0; i < 8; ++i) {
#pragma unroll
      for (int m = 1; m < 16; m <<= 1) rmax[i] = fmaxf(rmax[i], __shfl_xor(rmax[i], m));
      float m_new = fmaxf(m_run[i], rmax[i]);
      float sc = __expf(m_run[i] - m_new);
      m_run[i] = m_new;
      l_run[i] *= sc;
      accO[0][i] *= sc;
      accO[1][i] *= sc;
      float rs = 0.0f;
#pragma unroll
      for (int nt = 0; nt < 8; ++nt) {
        float p = __expf(s[nt][i] - m_new);
        s[nt][i] = p;
        rs += p;
      }
#pragma unroll
      for (int m = 1; m < 16; m <<= 1) rs += __shfl_xor(rs, m);
      l_run[i] += rs;
    }

    // write P (f16) to LDS in A-matrix row-major layout
#pragma unroll
    for (int nt = 0; nt < 8; ++nt)
#pragma unroll
      for (int i = 0; i < 8; ++i)
        sP[wid][(i + 8 * half) * 128 + nt * 16 + lc] = (_Float16)s[nt][i];
    __syncthreads();

    // O += P @ V  (K=128 in 4 steps of 32, 2 output d-tiles)
    v16h pf[4];
#pragma unroll
    for (int ks = 0; ks < 4; ++ks)
      pf[ks] = load_frag_a(&sP[wid][0] + ks * 32, 128, lane);
#pragma unroll
    for (int dt = 0; dt < 2; ++dt)
#pragma unroll
      for (int ks = 0; ks < 4; ++ks) {
        v16h bf = load_frag_b(sVt + dt * 16 * 128 + ks * 32, 128, lane);
        accO[dt] = __builtin_amdgcn_wmma_f32_16x16x32_f16(false, pf[ks], false, bf,
                                                          (short)0, accO[dt], false, false);
      }
    __syncthreads();
  }

  // finalize: out[b*N + qrow, h*32 + d] = O / l
#pragma unroll
  for (int dt = 0; dt < 2; ++dt)
#pragma unroll
    for (int i = 0; i < 8; ++i) {
      int qg = b * DIM_N + qrow0 + i + 8 * half;
      int col = h * DIM_HD + dt * 16 + lc;
      out[(size_t)qg * DIM_D + col] = (_Float16)(accO[dt][i] / l_run[i]);
    }
}

// ---------------- host orchestration ----------------
extern "C" void kernel_launch(void* const* d_in, const int* in_sizes, int n_in,
                              void* d_out, int out_size, void* d_ws, size_t ws_size,
                              hipStream_t stream) {
  const float* pixel  = (const float*)d_in[0];
  const float* conv_w = (const float*)d_in[1];
  const float* conv_b = (const float*)d_in[2];
  const float* pe_g   = (const float*)d_in[3];
  const float* pe_b   = (const float*)d_in[4];
  const float* n1g    = (const float*)d_in[5];
  const float* n1b    = (const float*)d_in[6];
  const float* qkv_w  = (const float*)d_in[7];
  const float* qkv_b  = (const float*)d_in[8];
  const float* proj_w = (const float*)d_in[9];
  const float* proj_b = (const float*)d_in[10];
  const float* rel_b  = (const float*)d_in[11];
  const float* n2g    = (const float*)d_in[12];
  const float* n2b    = (const float*)d_in[13];
  const float* w1     = (const float*)d_in[14];
  const float* b1     = (const float*)d_in[15];
  const float* w2     = (const float*)d_in[16];
  const float* b2     = (const float*)d_in[17];
  const float* fn_g   = (const float*)d_in[18];
  const float* fn_b   = (const float*)d_in[19];
  float* out = (float*)d_out;

  char* base = (char*)d_ws;
  size_t off = 0;
  auto alloc = [&](size_t bytes) -> void* {
    void* p = base + off;
    off = (off + bytes + 255) & ~(size_t)255;
    return p;
  };
  float*     x      = (float*)alloc((size_t)ROWS * DIM_D * 4);
  _Float16*  h16    = (_Float16*)alloc((size_t)ROWS * DIM_D * 2);
  _Float16*  attn16 = (_Float16*)alloc((size_t)ROWS * DIM_D * 2);
  _Float16*  g16    = (_Float16*)alloc((size_t)ROWS * 1536 * 2);
  size_t qkv_sz = (size_t)DIM_B * DIM_NH * DIM_N * DIM_HD;
  _Float16*  q16 = (_Float16*)alloc(qkv_sz * 2);
  _Float16*  k16 = (_Float16*)alloc(qkv_sz * 2);
  _Float16*  v16 = (_Float16*)alloc(qkv_sz * 2);
  size_t nqw = (size_t)DIM_DEPTH * 1152 * 384;
  size_t npw = (size_t)DIM_DEPTH * 384 * 384;
  size_t n1w = (size_t)DIM_DEPTH * 1536 * 384;
  size_t n2w = (size_t)DIM_DEPTH * 384 * 1536;
  _Float16* wq16 = (_Float16*)alloc(nqw * 2);
  _Float16* wp16 = (_Float16*)alloc(npw * 2);
  _Float16* w116 = (_Float16*)alloc(n1w * 2);
  _Float16* w216 = (_Float16*)alloc(n2w * 2);

  // weight conversion fp32 -> f16
  cvt_f16_kernel<<<(int)((nqw + 255) / 256), 256, 0, stream>>>(qkv_w,  wq16, (int)nqw);
  cvt_f16_kernel<<<(int)((npw + 255) / 256), 256, 0, stream>>>(proj_w, wp16, (int)npw);
  cvt_f16_kernel<<<(int)((n1w + 255) / 256), 256, 0, stream>>>(w1,     w116, (int)n1w);
  cvt_f16_kernel<<<(int)((n2w + 255) / 256), 256, 0, stream>>>(w2,     w216, (int)n2w);

  // patch embed + its LayerNorm (in place)
  patch_embed_kernel<<<ROWS, 128, 0, stream>>>(pixel, conv_w, conv_b, x);
  ln_kernel<float><<<ROWS, 128, 0, stream>>>(x, pe_g, pe_b, x);

  const float qscale = 0.17677669529663688f;   // 1/sqrt(32)
  for (int dep = 0; dep < DIM_DEPTH; ++dep) {
    ln_kernel<_Float16><<<ROWS, 128, 0, stream>>>(x, n1g + dep * 384, n1b + dep * 384, h16);
    gemm_kernel<0><<<dim3(9, 32), 256, 0, stream>>>(
        h16, wq16 + (size_t)dep * 1152 * 384, qkv_b + dep * 1152,
        nullptr, nullptr, nullptr, q16, k16, v16, ROWS, 384, 1152, qscale);
    attn_kernel<<<dim3(16, 12, 4), 128, 0, stream>>>(
        q16, k16, v16, rel_b + (size_t)dep * DIM_TABLE * DIM_NH, attn16);
    gemm_kernel<1><<<dim3(3, 32), 256, 0, stream>>>(
        attn16, wp16 + (size_t)dep * 384 * 384, proj_b + dep * 384,
        x, x, nullptr, nullptr, nullptr, nullptr, ROWS, 384, 384, 0.0f);
    ln_kernel<_Float16><<<ROWS, 128, 0, stream>>>(x, n2g + dep * 384, n2b + dep * 384, h16);
    gemm_kernel<2><<<dim3(12, 32), 256, 0, stream>>>(
        h16, w116 + (size_t)dep * 1536 * 384, b1 + dep * 1536,
        nullptr, nullptr, g16, nullptr, nullptr, nullptr, ROWS, 384, 1536, 0.0f);
    gemm_kernel<1><<<dim3(3, 32), 256, 0, stream>>>(
        g16, w216 + (size_t)dep * 384 * 1536, b2 + dep * 384,
        x, x, nullptr, nullptr, nullptr, nullptr, ROWS, 1536, 384, 0.0f);
  }
  ln_kernel<float><<<ROWS, 128, 0, stream>>>(x, fn_g, fn_b, out);
}
